// W6Linear_81638738362815
// MI455X (gfx1250) — matrix operand order repaired and checked by reading, not compiled, not run on hardware
//
#include <hip/hip_runtime.h>

typedef __bf16 bf16;
typedef __attribute__((ext_vector_type(16))) __bf16 v16bf;
typedef __attribute__((ext_vector_type(8)))  float  v8f;
typedef __attribute__((ext_vector_type(4)))  int    v4i;

#define M_DIM 16384
#define K_DIM 4096
#define N_DIM 4096

#if defined(__AMDGCN__) && __has_builtin(__builtin_amdgcn_global_load_async_to_lds_b128) && __has_builtin(__builtin_amdgcn_s_wait_asynccnt)
#define USE_ASYNC_LDS 1
typedef __attribute__((address_space(1))) v4i glob_v4i;  // global raw int4
typedef __attribute__((address_space(3))) v4i lds_v4i;   // LDS raw int4
#else
#define USE_ASYNC_LDS 0
#endif

// ---------------------------------------------------------------------------
// Pass 1: unpack 6-bit weights, fold zero-point/scale, emit bf16 W[N][K].
// ---------------------------------------------------------------------------
__global__ __launch_bounds__(256) void dequant_kernel(
    const int* __restrict__ q, const float* __restrict__ scale,
    const float* __restrict__ zp, bf16* __restrict__ wb) {
  int g  = blockIdx.x * 256 + threadIdx.x;   // 4096 * 1024 groups
  int n  = g >> 10;
  int gi = g & 1023;
  const int* p = q + (size_t)n * 3072 + gi * 3;
  int p0 = p[0], p1 = p[1], p2 = p[2];
  float s = scale[n];
  float z = zp[n];
  union { bf16 h[4]; uint2 u; } pk;
  pk.h[0] = (bf16)(((float)(p0 & 63)        - z) * s);
  pk.h[1] = (bf16)(((float)((p0 >> 6) & 63) - z) * s);
  pk.h[2] = (bf16)(((float)(p1 & 63)        - z) * s);
  pk.h[3] = (bf16)(((float)(p2 & 63)        - z) * s);
  *(uint2*)(wb + (size_t)n * K_DIM + gi * 4) = pk.u;
}

// ---------------------------------------------------------------------------
// Pass 2: convert activations f32 -> bf16
// ---------------------------------------------------------------------------
__global__ __launch_bounds__(256) void convx_kernel(
    const float* __restrict__ x, bf16* __restrict__ xb) {
  size_t i4 = (size_t)blockIdx.x * 256 + threadIdx.x;  // M*K/4 threads
  float4 v = *(const float4*)(x + i4 * 4);
  union { bf16 h[4]; uint2 u; } pk;
  pk.h[0] = (bf16)v.x;
  pk.h[1] = (bf16)v.y;
  pk.h[2] = (bf16)v.z;
  pk.h[3] = (bf16)v.w;
  *(uint2*)(xb + i4 * 4) = pk.u;
}

// ---------------------------------------------------------------------------
// Pass 3: GEMM  out[M][N] = Xbf16[M][K] * Wbf16[N][K]^T + bias
// Block = 256 threads = 8 waves. Block tile 128(M) x 128(N), K step 32,
// double-buffered LDS with CDNA5 async global->LDS copies (ASYNCcnt).
// Wave tile 32x64 = 2x4 v_wmma_f32_16x16x32_bf16 tiles.
// ---------------------------------------------------------------------------
__global__ __launch_bounds__(256) void gemm_kernel(
    const bf16* __restrict__ A,   // [M][K] bf16
    const bf16* __restrict__ B,   // [N][K] bf16
    const float* __restrict__ bias,
    float* __restrict__ out) {
  __shared__ __align__(16) bf16 As[2][128][40];  // +8 pad: bank stagger
  __shared__ __align__(16) bf16 Bs[2][128][40];

  const int tid  = threadIdx.x;
  const int lane = tid & 31;
  const int wave = tid >> 5;
  const int wm   = wave >> 1;           // 0..3  (M sub-tile, 32 rows each)
  const int wn   = wave & 1;            // 0..1  (N sub-tile, 64 cols each)
  const int bm   = blockIdx.y * 128;
  const int bn   = blockIdx.x * 128;

  // staging: each thread moves 16 A elems + 16 B elems per K-step (2+2 b128)
  const int sr  = tid >> 1;             // 0..127 row
  const int scc = (tid & 1) * 16;       // 0 / 16

  const bf16* aGlob = A + (size_t)(bm + sr) * K_DIM + scc;
  const bf16* bGlob = B + (size_t)(bn + sr) * K_DIM + scc;

  // fragment gather coordinates (CDNA5 WMMA VGPR layouts)
  const int arow = wm * 32 + (lane & 15);
  const int akb  = (lane >> 4) * 8;
  const int brow = wn * 64 + (lane & 15);
  const int bkb  = (lane >> 4) * 8;

  v8f acc[2][4] = {};
  union Frag { v16bf v; uint4 q[2]; };

  constexpr int KT = K_DIM / 32;        // 128 K-steps

#if USE_ASYNC_LDS
  #define STAGE(buf, k0)                                                      \
    do {                                                                      \
      __builtin_amdgcn_global_load_async_to_lds_b128(                         \
          (glob_v4i*)(aGlob + (k0)),     (lds_v4i*)&As[buf][sr][scc],     0, 0);\
      __builtin_amdgcn_global_load_async_to_lds_b128(                         \
          (glob_v4i*)(aGlob + (k0) + 8), (lds_v4i*)&As[buf][sr][scc + 8], 0, 0);\
      __builtin_amdgcn_global_load_async_to_lds_b128(                         \
          (glob_v4i*)(bGlob + (k0)),     (lds_v4i*)&Bs[buf][sr][scc],     0, 0);\
      __builtin_amdgcn_global_load_async_to_lds_b128(                         \
          (glob_v4i*)(bGlob + (k0) + 8), (lds_v4i*)&Bs[buf][sr][scc + 8], 0, 0);\
    } while (0)
  #define STAGE_WAIT() __builtin_amdgcn_s_wait_asynccnt(0)
#endif

  // one K-step of compute from LDS buffer `buf`
  auto compute = [&](int buf) {
    Frag a[2], b[4];
#pragma unroll
    for (int mi = 0; mi < 2; ++mi) {
      a[mi].q[0] = *(const uint4*)&As[buf][arow + mi * 16][akb];
      a[mi].q[1] = *(const uint4*)&As[buf][arow + mi * 16][akb + 16];
    }
#pragma unroll
    for (int ni = 0; ni < 4; ++ni) {
      b[ni].q[0] = *(const uint4*)&Bs[buf][brow + ni * 16][bkb];
      b[ni].q[1] = *(const uint4*)&Bs[buf][brow + ni * 16][bkb + 16];
    }
#pragma unroll
    for (int mi = 0; mi < 2; ++mi)
#pragma unroll
      for (int ni = 0; ni < 4; ++ni)
        acc[mi][ni] = __builtin_amdgcn_wmma_f32_16x16x32_bf16(
            false, a[mi].v, false, b[ni].v, (short)0, acc[mi][ni], false, false);
  };

#if USE_ASYNC_LDS
  // prologue: fill buffer 0
  STAGE(0, 0);
  STAGE_WAIT();
  __syncthreads();
  for (int kt = 0; kt < KT - 1; ++kt) {
    const int cur = kt & 1;
    const int nxt = cur ^ 1;
    STAGE(nxt, (kt + 1) * 32);          // async copy next tile (no VGPR trip)
    compute(cur);                       // WMMA on current tile
    STAGE_WAIT();                       // drain ASYNCcnt
    __syncthreads();
  }
  compute((KT - 1) & 1);
#else
  // fallback: register-staged double buffering
  {
    uint4 sa0 = *(const uint4*)(aGlob);
    uint4 sa1 = *(const uint4*)(aGlob + 8);
    uint4 sb0 = *(const uint4*)(bGlob);
    uint4 sb1 = *(const uint4*)(bGlob + 8);
    *(uint4*)&As[0][sr][scc]     = sa0;
    *(uint4*)&As[0][sr][scc + 8] = sa1;
    *(uint4*)&Bs[0][sr][scc]     = sb0;
    *(uint4*)&Bs[0][sr][scc + 8] = sb1;
  }
  __syncthreads();
  for (int kt = 0; kt < KT - 1; ++kt) {
    const int cur = kt & 1;
    const int nxt = cur ^ 1;
    const int k0 = (kt + 1) * 32;
    uint4 sa0 = *(const uint4*)(aGlob + k0);        // issue early
    uint4 sa1 = *(const uint4*)(aGlob + k0 + 8);
    uint4 sb0 = *(const uint4*)(bGlob + k0);
    uint4 sb1 = *(const uint4*)(bGlob + k0 + 8);
    compute(cur);                                   // overlap with loads
    *(uint4*)&As[nxt][sr][scc]     = sa0;
    *(uint4*)&As[nxt][sr][scc + 8] = sa1;
    *(uint4*)&Bs[nxt][sr][scc]     = sb0;
    *(uint4*)&Bs[nxt][sr][scc + 8] = sb1;
    __syncthreads();
  }
  compute((KT - 1) & 1);
#endif

  // ---- epilogue: add bias, store f32 ----
  const int rbase = bm + wm * 32 + ((lane >> 4) * 8);
#pragma unroll
  for (int ni = 0; ni < 4; ++ni) {
    const int col = bn + wn * 64 + ni * 16 + (lane & 15);
    const float bv = bias[col];
#pragma unroll
    for (int v = 0; v < 8; ++v) {
      out[(size_t)(rbase + v)      * N_DIM + col] = acc[0][ni][v] + bv;
      out[(size_t)(rbase + 16 + v) * N_DIM + col] = acc[1][ni][v] + bv;
    }
  }
}

// ---------------------------------------------------------------------------
extern "C" void kernel_launch(void* const* d_in, const int* in_sizes, int n_in,
                              void* d_out, int out_size, void* d_ws, size_t ws_size,
                              hipStream_t stream) {
  const float* x    = (const float*)d_in[0];   // (8,2048,4096) f32
  const int*   qw   = (const int*)  d_in[1];   // (4096,3072)   i32
  const float* sc   = (const float*)d_in[2];   // (4096,1)
  const float* zp   = (const float*)d_in[3];   // (4096,1)
  const float* bias = (const float*)d_in[4];   // (4096,)
  float* out        = (float*)d_out;           // (8,2048,4096) f32

  bf16* wb = (bf16*)d_ws;                      // 4096*4096 bf16 = 32 MB
  bf16* xb = wb + (size_t)N_DIM * K_DIM;       // 16384*4096 bf16 = 128 MB

  dequant_kernel<<<(N_DIM * 1024) / 256, 256, 0, stream>>>(qw, sc, zp, wb);
  convx_kernel<<<(int)(((size_t)M_DIM * K_DIM / 4) / 256), 256, 0, stream>>>(x, xb);
  dim3 grid(N_DIM / 128, M_DIM / 128);
  gemm_kernel<<<grid, 256, 0, stream>>>(xb, wb, bias, out);
}
